// RTDETRTransformer_64046552318319
// MI455X (gfx1250) — matrix operand order repaired and checked by reading, not compiled, not run on hardware
//
#include <hip/hip_runtime.h>
#include <hip/hip_bf16.h>

typedef __attribute__((ext_vector_type(16))) _Float16 v16h;
typedef __attribute__((ext_vector_type(8)))  _Float16 v8h;
typedef __attribute__((ext_vector_type(4)))  float    v4f;
typedef __attribute__((ext_vector_type(8)))  float    v8f;
typedef int v4i_vs __attribute__((vector_size(16)));   // matches builtin proto

#if __has_builtin(__builtin_amdgcn_global_load_async_to_lds_b128) && \
    __has_builtin(__builtin_amdgcn_s_wait_asynccnt)
#define USE_ASYNC_LDS 1
#endif

// ---------------------------------------------------------------------------
// Problem constants
// ---------------------------------------------------------------------------
#define BS   32
#define LQ   300
#define CDIM 256
#define NH   8
#define NL   3
#define NP   4
#define HD   32
#define LV   8400            // 80*80 + 40*40 + 20*20
#define MQ   (BS * LQ)       // 9600
#define MV   (BS * LV)       // 268800
#define NCAT 320             // 192 (off) + 96 (att) padded to 320

// ---------------------------------------------------------------------------
// Kernel 0: weight conversion to f16, transposed to [n][k] (k contiguous)
// ---------------------------------------------------------------------------
__global__ void prep_weights(const float* __restrict__ Wv,
                             const float* __restrict__ Woff,
                             const float* __restrict__ Watt,
                             const float* __restrict__ Wout,
                             const float* __restrict__ boff,
                             const float* __restrict__ batt,
                             _Float16* __restrict__ WvT,
                             _Float16* __restrict__ WcatT,
                             float*    __restrict__ bcat,
                             _Float16* __restrict__ WoutT) {
  int t = blockIdx.x * blockDim.x + threadIdx.x;
  int stride = gridDim.x * blockDim.x;
  for (int i = t; i < CDIM * CDIM; i += stride) {
    int n = i >> 8, k = i & 255;                 // dst[n][k] = src[k][n]
    WvT[i]   = (_Float16)Wv[k * CDIM + n];
    WoutT[i] = (_Float16)Wout[k * CDIM + n];
  }
  for (int i = t; i < NCAT * CDIM; i += stride) {
    int n = i >> 8, k = i & 255;
    float v = 0.0f;
    if (n < 192)      v = Woff[k * 192 + n];
    else if (n < 288) v = Watt[k * 96 + (n - 192)];
    WcatT[i] = (_Float16)v;
  }
  for (int i = t; i < NCAT; i += stride) {
    float v = 0.0f;
    if (i < 192)      v = boff[i];
    else if (i < 288) v = batt[i - 192];
    bcat[i] = v;
  }
}

// ---------------------------------------------------------------------------
// WMMA GEMM: D[M,N] = A[M,K](fp32) * Bt[N,K](f16, transposed) + bias[N]
//   block = 64 x N strip (A read exactly once from HBM), 8 waves,
//   wave w: m tile (w&3), n tiles [(w>>2)*NT, (w>>2)*NT + NT)
//   NT = N/32 tiles per wave (8 for N=256, 10 for N=320)
//   MODE 0: store fp32, row stride N              (offatt)
//   MODE 1: store f16 permuted [b][h][pix][ch]    (value projection)
//   MODE 2: store fp32, row stride N -> d_out     (output projection)
// ---------------------------------------------------------------------------
#define BM 64
#define KT 32
#define LDAH 40   // padded LDS stride in halves (80B rows, 16B-aligned chunks)

template <int NT, int MODE>
__global__ __launch_bounds__(256) void gemm_f16w(
    const float* __restrict__ A, const _Float16* __restrict__ Bt,
    const float* __restrict__ bias, void* __restrict__ Dout, int K) {
  constexpr int N = NT * 32;
  __shared__ _Float16 sA[BM * LDAH];   // [row][k]
  __shared__ _Float16 sB[N * LDAH];    // [col][k]

  const int tid  = threadIdx.x;
  const int lane = tid & 31;
  const int wv   = tid >> 5;
  const int m0   = blockIdx.x * BM;

  const int m_blk  = wv & 3;   // 0..3
  const int n_half = wv >> 2;  // 0..1

  v8f acc[NT] = {};

  // A fragment: lane 0-15 -> row, K {0..7,16..23}; lane 16-31 -> K {8..15,24..31}
  const int row_a = m_blk * 16 + (lane & 15);
  const int kbA   = (lane >> 4) * 8;
  // B fragment: lane 0-15 -> col, K {0..7,16..23}; lane 16-31 -> K {8..15,24..31}
  const int nl    = lane & 15;
  const int kbB   = (lane >> 4) * 8;

  // A staging addressing: thread -> (row r = tid>>2, 8 consecutive k at (tid&3)*8)
  const int ar = tid >> 2;
  const int ac = (tid & 3) * 8;

  for (int kt = 0; kt < K; kt += KT) {
    // ---- stage A tile 64x32 (fp32 -> f16), coalesced 2x b128 loads ----
    {
      const float* src = A + (size_t)(m0 + ar) * K + kt + ac;
      v4f f0 = *(const v4f*)src;
      v4f f1 = *(const v4f*)(src + 4);
      v8h hv;
#pragma unroll
      for (int i = 0; i < 4; ++i) { hv[i] = (_Float16)f0[i]; hv[i + 4] = (_Float16)f1[i]; }
      *(v8h*)&sA[ar * LDAH + ac] = hv;
    }
    // ---- stage B tile Nx32 as [n][k]: 16B chunks, async copy to LDS ----
#pragma unroll
    for (int i = 0; i < NT / 2; ++i) {
      int c  = tid + i * 256;       // chunk id in [0, N*4)
      int n  = c >> 2;
      int co = (c & 3) * 8;
      const _Float16* gsrc = Bt + (size_t)n * CDIM + kt + co;
      _Float16*       ldst = &sB[n * LDAH + co];
#ifdef USE_ASYNC_LDS
      __builtin_amdgcn_global_load_async_to_lds_b128(
          (v4i_vs*)gsrc, (v4i_vs*)ldst, 0, 0);
#else
      *(v8h*)ldst = *(const v8h*)gsrc;
#endif
    }
#ifdef USE_ASYNC_LDS
    __builtin_amdgcn_s_wait_asynccnt(0);
#endif
    __syncthreads();

    // ---- fragments + WMMA ----
    v16h af;
    {
      v8h lo = *(const v8h*)&sA[row_a * LDAH + kbA];
      v8h hi = *(const v8h*)&sA[row_a * LDAH + kbA + 16];
#pragma unroll
      for (int i = 0; i < 8; ++i) { af[i] = lo[i]; af[i + 8] = hi[i]; }
    }
#pragma unroll
    for (int j = 0; j < NT; ++j) {
      const int col = (n_half * NT + j) * 16 + nl;
      v16h bf;
      v8h lo = *(const v8h*)&sB[col * LDAH + kbB];
      v8h hi = *(const v8h*)&sB[col * LDAH + kbB + 16];
#pragma unroll
      for (int i = 0; i < 8; ++i) { bf[i] = lo[i]; bf[i + 8] = hi[i]; }
      acc[j] = __builtin_amdgcn_wmma_f32_16x16x32_f16(false, af, false, bf,
                                                      (short)0, acc[j],
                                                      false, false);
    }
    __syncthreads();
  }

  // ---- epilogue: element i -> (M = i + 8*(lane>>4), N = lane&15) ----
  const int mloc = m_blk * 16 + 8 * (lane >> 4);
#pragma unroll
  for (int j = 0; j < NT; ++j) {
    const int n = (n_half * NT + j) * 16 + nl;
    const float bn = bias[n];
#pragma unroll
    for (int i = 0; i < 8; ++i) {
      int m = m0 + mloc + i;
      float val = acc[j][i] + bn;
      if (MODE == 1) {
        // m = b*LV + pix ; n = h*HD + ch ; -> v_h[b][h][pix][ch]
        unsigned um = (unsigned)m;
        unsigned b = um / (unsigned)LV;
        unsigned pix = um - b * (unsigned)LV;
        size_t dst = ((size_t)(b * NH + (unsigned)(n >> 5)) * LV + pix) * HD +
                     (unsigned)(n & 31);
        ((_Float16*)Dout)[dst] = (_Float16)val;
      } else {
        ((float*)Dout)[(size_t)m * N + n] = val;
      }
    }
  }
}

// ---------------------------------------------------------------------------
// Kernel 2: softmax over 12 attention logits + sampling locations
//   one thread per (b,q,h); g = (b*LQ + q)*NH + h
// ---------------------------------------------------------------------------
__global__ void loc_att_kernel(const float* __restrict__ offatt,
                               const float* __restrict__ refp,
                               float* __restrict__ loc,
                               float* __restrict__ attw) {
  int g = blockIdx.x * blockDim.x + threadIdx.x;
  if (g >= BS * LQ * NH) return;
  int h  = g & 7;
  int bq = g >> 3;

  const float* row = offatt + (size_t)bq * NCAT;
  const float* rp  = refp + (size_t)bq * (NL * 4);

  float lg[NL * NP];
  float mx = -1e30f;
#pragma unroll
  for (int i = 0; i < NL * NP; ++i) {
    lg[i] = row[192 + h * 12 + i];
    mx = fmaxf(mx, lg[i]);
  }
  float s = 0.0f;
#pragma unroll
  for (int i = 0; i < NL * NP; ++i) { lg[i] = __expf(lg[i] - mx); s += lg[i]; }
  float inv = 1.0f / s;

  float* lo = loc  + (size_t)g * (NL * NP * 2);
  float* aw = attw + (size_t)g * (NL * NP);
#pragma unroll
  for (int l = 0; l < NL; ++l) {
    float rx = rp[l * 4 + 0], ry = rp[l * 4 + 1];
    float rw = rp[l * 4 + 2], rh = rp[l * 4 + 3];
#pragma unroll
    for (int p = 0; p < NP; ++p) {
      int i = l * NP + p;
      float ox = row[((h * NL + l) * NP + p) * 2 + 0];
      float oy = row[((h * NL + l) * NP + p) * 2 + 1];
      lo[i * 2 + 0] = rx + ox * 0.125f * rw;   // off/P * wh * 0.5
      lo[i * 2 + 1] = ry + oy * 0.125f * rh;
      aw[i] = lg[i] * inv;
    }
  }
}

// ---------------------------------------------------------------------------
// Kernel 3: bilinear sampling + attention-weighted reduce
//   one wave32 per (b,q,h); lane = channel (HD == 32)
// ---------------------------------------------------------------------------
__global__ __launch_bounds__(256) void sample_kernel(
    const _Float16* __restrict__ vh, const float* __restrict__ loc,
    const float* __restrict__ attw, float* __restrict__ mid) {
  int wid  = (blockIdx.x << 3) + (threadIdx.x >> 5);   // g index
  int lane = threadIdx.x & 31;
  int h  = wid & 7;
  int bq = wid >> 3;
  int b  = bq / LQ;

  const _Float16* vb = vh + (size_t)(b * NH + h) * LV * HD;
  const float* lo = loc  + (size_t)wid * (NL * NP * 2);
  const float* aw = attw + (size_t)wid * (NL * NP);

  const int Hs[NL] = {80, 40, 20};
  const int Ws[NL] = {80, 40, 20};
  const int st[NL] = {0, 6400, 8000};

  float acc = 0.0f;
#pragma unroll
  for (int l = 0; l < NL; ++l) {
    const int W = Ws[l], H = Hs[l], base = st[l];
#pragma unroll
    for (int p = 0; p < NP; ++p) {
      int i = l * NP + p;
      float a  = aw[i];
      float px = lo[i * 2 + 0] * (float)W - 0.5f;
      float py = lo[i * 2 + 1] * (float)H - 0.5f;
      float x0f = floorf(px), y0f = floorf(py);
      int x0 = (int)x0f, y0 = (int)y0f;
      float wx1 = px - x0f, wy1 = py - y0f;
      float wx0 = 1.0f - wx1, wy0 = 1.0f - wy1;

      if (x0 >= 0 && x0 < W && y0 >= 0 && y0 < H)
        acc += a * wx0 * wy0 *
               (float)vb[(size_t)(base + y0 * W + x0) * HD + lane];
      if (x0 + 1 >= 0 && x0 + 1 < W && y0 >= 0 && y0 < H)
        acc += a * wx1 * wy0 *
               (float)vb[(size_t)(base + y0 * W + x0 + 1) * HD + lane];
      if (x0 >= 0 && x0 < W && y0 + 1 >= 0 && y0 + 1 < H)
        acc += a * wx0 * wy1 *
               (float)vb[(size_t)(base + (y0 + 1) * W + x0) * HD + lane];
      if (x0 + 1 >= 0 && x0 + 1 < W && y0 + 1 >= 0 && y0 + 1 < H)
        acc += a * wx1 * wy1 *
               (float)vb[(size_t)(base + (y0 + 1) * W + x0 + 1) * HD + lane];
    }
  }
  mid[(size_t)bq * CDIM + h * HD + lane] = acc;
}

// ---------------------------------------------------------------------------
// Launch
// ---------------------------------------------------------------------------
extern "C" void kernel_launch(void* const* d_in, const int* in_sizes, int n_in,
                              void* d_out, int out_size, void* d_ws, size_t ws_size,
                              hipStream_t stream) {
  const float* query = (const float*)d_in[0];
  const float* refp  = (const float*)d_in[1];
  const float* value = (const float*)d_in[2];
  const float* Wv    = (const float*)d_in[3];
  const float* bv    = (const float*)d_in[4];
  const float* Woff  = (const float*)d_in[5];
  const float* boff  = (const float*)d_in[6];
  const float* Watt  = (const float*)d_in[7];
  const float* batt  = (const float*)d_in[8];
  const float* Wout  = (const float*)d_in[9];
  const float* bout  = (const float*)d_in[10];
  float* out = (float*)d_out;

  char* ws = (char*)d_ws;
  size_t off = 0;
  auto take = [&](size_t bytes) -> void* {
    off = (off + 255) & ~(size_t)255;
    void* p = ws + off;
    off += bytes;
    return p;
  };

  _Float16* WvT    = (_Float16*)take((size_t)CDIM * CDIM * 2);
  _Float16* WcatT  = (_Float16*)take((size_t)NCAT * CDIM * 2);
  float*    bcat   = (float*)   take((size_t)NCAT * 4);
  _Float16* WoutT  = (_Float16*)take((size_t)CDIM * CDIM * 2);
  _Float16* v_h    = (_Float16*)take((size_t)MV * CDIM * 2);       // ~137 MB
  float*    offatt = (float*)   take((size_t)MQ * NCAT * 4);       // ~12 MB
  float*    loc    = (float*)   take((size_t)MQ * NH * NL * NP * 2 * 4);
  float*    attw   = (float*)   take((size_t)MQ * NH * NL * NP * 4);
  float*    mid    = (float*)   take((size_t)MQ * CDIM * 4);

  // 0) weights -> f16, transposed [n][k] (+ concat/pad Woff|Watt)
  prep_weights<<<320, 256, 0, stream>>>(Wv, Woff, Watt, Wout, boff, batt,
                                        WvT, WcatT, bcat, WoutT);

  // 1) v = value @ Wv + bv  -> f16, permuted [b][h][pix][ch]; full-N strip
  gemm_f16w<8, 1><<<MV / BM, 256, 0, stream>>>(value, WvT, bv, (void*)v_h, CDIM);

  // 2) offatt = query @ [Woff|Watt] + [boff|batt]  (N padded to 320)
  gemm_f16w<10, 0><<<MQ / BM, 256, 0, stream>>>(query, WcatT, bcat,
                                                (void*)offatt, CDIM);

  // 3) softmax + sampling locations
  loc_att_kernel<<<(BS * LQ * NH + 255) / 256, 256, 0, stream>>>(
      offatt, refp, loc, attw);

  // 4) bilinear sample + weighted sum  (wave per (b,q,h))
  sample_kernel<<<(BS * LQ * NH) / 8, 256, 0, stream>>>(v_h, loc, attw, mid);

  // 5) out = mid @ Wout + bout
  gemm_f16w<8, 2><<<MQ / BM, 256, 0, stream>>>(mid, WoutT, bout, (void*)out,
                                               CDIM);
}
